// OptimizedLegalMultiHeadAttention_84456236908950
// MI455X (gfx1250) — compile-verified, matrix operand order checked by reading, not run on hardware
//
#include <hip/hip_runtime.h>
#include <hip/hip_bf16.h>

#define DEV __device__ __forceinline__

// ---- CDNA5 async-to-LDS path (ASYNCcnt), guarded so either toolchain compiles
#if defined(__has_builtin)
#  if __has_builtin(__builtin_amdgcn_global_load_async_to_lds_b128) && \
      __has_builtin(__builtin_amdgcn_s_wait_asynccnt)
#    define HAVE_ASYNC_LDS 1
#  endif
#endif
#ifndef HAVE_ASYNC_LDS
#  define HAVE_ASYNC_LDS 0
#endif

typedef __attribute__((ext_vector_type(4)))  float  f32x4;
typedef __attribute__((ext_vector_type(8)))  float  v8f;
typedef __attribute__((ext_vector_type(4)))  __bf16 bf16x4;
typedef __attribute__((ext_vector_type(8)))  __bf16 bf16x8;
typedef __attribute__((ext_vector_type(16))) __bf16 bf16x16;

// exact parameter type reported by the compiler for the async builtin
typedef __attribute__((__vector_size__(4 * sizeof(int)))) int v4i_t;

// Problem constants (match reference)
constexpr int   Bn   = 2;             // batch
constexpr int   Sq   = 2048;          // sequence
constexpr int   Dm   = 1024;          // d_model
constexpr int   NHs  = 4;             // sub-heads per named head
constexpr int   Hn   = 4;             // named heads
constexpr int   DKd  = 256;           // per-named-head projection dim
constexpr int   DSd  = 64;            // sub-head dim
constexpr int   BS   = Bn * Sq;       // 4096
constexpr int   NCMB = Hn * Bn * NHs; // 32 (h,b,n) combos
constexpr float SCALE = 0.0625f;      // 1/sqrt(256)

constexpr int LDST = 56;              // LDS row stride (elems): padded 32->56, 112B (16B aligned)

// ---------------------------------------------------------------------------
// async copy helpers
// ---------------------------------------------------------------------------
template <int N> DEV void wait_async() {
#if HAVE_ASYNC_LDS
    __builtin_amdgcn_s_wait_asynccnt(N);
#endif
}

DEV void async_copy16(const __bf16* g, __bf16* l) {
#if HAVE_ASYNC_LDS
    __builtin_amdgcn_global_load_async_to_lds_b128(
        (v4i_t*)(g),
        (__attribute__((address_space(3))) v4i_t*)(l),
        0, 0);
#else
    *(bf16x8*)l = *(const bf16x8*)g;   // sync fallback: global b128 -> ds_store_b128
#endif
}

// ---------------------------------------------------------------------------
// WMMA fragment loaders.
// 16-bit A-matrix 16x32 layout (ISA 7.12.2): lane L -> row m = L%16, half = L/16,
// holds k in [8h, 8h+8) and [16+8h, 16+8h+8)  => two 16B loads per lane.
// B operand supplied as B^T rows with the identical layout (n = L%16).
// ---------------------------------------------------------------------------
DEV bf16x16 load_frag_bf16(const __bf16* __restrict__ base, int ld, int k0) {
    int lane = threadIdx.x & 31;
    int r    = lane & 15;
    int half = lane >> 4;
    const __bf16* p = base + (size_t)r * ld + k0 + half * 8;
    union { bf16x16 f; bf16x8 h[2]; } u;
    u.h[0] = *(const bf16x8*)(p);
    u.h[1] = *(const bf16x8*)(p + 16);
    return u.f;
}

DEV bf16x16 frag_lds(const __bf16* base /* row0 of 16-row tile, stride LDST */) {
    int lane = threadIdx.x & 31;
    int r    = lane & 15;
    int half = lane >> 4;
    const __bf16* p = base + r * LDST + half * 8;
    union { bf16x16 f; bf16x8 h[2]; } u;
    u.h[0] = *(const bf16x8*)(p);
    u.h[1] = *(const bf16x8*)(p + 16);
    return u.f;
}

DEV bf16x16 load_frag_f32_cvt(const float* __restrict__ base, int ld, int k0) {
    int lane = threadIdx.x & 31;
    int r    = lane & 15;
    int half = lane >> 4;
    const float* p = base + (size_t)r * ld + k0 + half * 8;
    f32x4 c0 = *(const f32x4*)(p);
    f32x4 c1 = *(const f32x4*)(p + 4);
    f32x4 c2 = *(const f32x4*)(p + 16);
    f32x4 c3 = *(const f32x4*)(p + 20);
    union { bf16x16 f; __bf16 e[16]; } u;
#pragma unroll
    for (int i = 0; i < 4; ++i) {
        u.e[i]      = (__bf16)c0[i];
        u.e[4 + i]  = (__bf16)c1[i];
        u.e[8 + i]  = (__bf16)c2[i];
        u.e[12 + i] = (__bf16)c3[i];
    }
    return u.f;
}

DEV v8f wmma_bf16(bf16x16 a, bf16x16 b, v8f c) {
    return __builtin_amdgcn_wmma_f32_16x16x32_bf16(
        /*neg_a=*/false, a, /*neg_b=*/false, b,
        /*c_mod=*/(short)0, c, /*reuse_a=*/false, /*reuse_b=*/false);
}

// ---------------------------------------------------------------------------
// 0) fp32 -> bf16 elementwise convert (4 elems/thread)
// ---------------------------------------------------------------------------
__global__ __launch_bounds__(256) void cvt_f32_to_bf16(const float* __restrict__ in,
                                                       __bf16* __restrict__ out, int n4) {
    int i = blockIdx.x * blockDim.x + threadIdx.x;
    if (i >= n4) return;
    f32x4 v = ((const f32x4*)in)[i];
    bf16x4 o;
#pragma unroll
    for (int k = 0; k < 4; ++k) o[k] = (__bf16)v[k];
    ((bf16x4*)out)[i] = o;
}

// ---------------------------------------------------------------------------
// 1) mask [B,S,S] int32 -> 1-bit/elem bitmap (1 MB; stays L2-resident)
// ---------------------------------------------------------------------------
__global__ __launch_bounds__(256) void pack_mask(const int* __restrict__ mask,
                                                 unsigned* __restrict__ mbits) {
    int idx = blockIdx.x * blockDim.x + threadIdx.x;   // one element per thread
    unsigned pred = (mask[idx] != 0) ? 1u : 0u;
#if defined(__has_builtin) && __has_builtin(__builtin_amdgcn_ballot_w32)
    unsigned bal = __builtin_amdgcn_ballot_w32(pred != 0);
#else
    unsigned bal = (unsigned)__ballot(pred);
#endif
    if ((threadIdx.x & 31) == 0) mbits[idx >> 5] = bal;
}

// ---------------------------------------------------------------------------
// 2) transpose + convert: out[c*rows + r] = (bf16) in[r*cols + c]
// ---------------------------------------------------------------------------
__global__ __launch_bounds__(256) void transpose_cvt(const float* __restrict__ in,
                                                     __bf16* __restrict__ out,
                                                     int rows, int cols,
                                                     long inHeadStride, long outHeadStride) {
    __shared__ float tile[32][33];
    const float* ip = in  + (size_t)blockIdx.z * inHeadStride;
    __bf16*      op = out + (size_t)blockIdx.z * outHeadStride;
    int c0 = blockIdx.x * 32, r0 = blockIdx.y * 32;
    int tx = threadIdx.x, ty = threadIdx.y;
#pragma unroll
    for (int i = 0; i < 32; i += 8)
        tile[ty + i][tx] = ip[(size_t)(r0 + ty + i) * cols + (c0 + tx)];
    __syncthreads();
#pragma unroll
    for (int i = 0; i < 32; i += 8)
        op[(size_t)(c0 + ty + i) * rows + (r0 + tx)] = (__bf16)tile[tx][ty + i];
}

// ---------------------------------------------------------------------------
// 3) bf16 GEMM, LDS double-buffered (async-to-LDS when available):
//    C = A[M,K] * B (given as Bt[N,K] row-major) + bias[N]
//    Block 8 waves: tile 128x64, k-step 32.
//    LAYOUT 0: C bf16 [M,N] | 1: Qh/Kh head-split | 2: VhT head-split-T | 3: C fp32 [M,N]
// ---------------------------------------------------------------------------
template <int LAYOUT>
__global__ __launch_bounds__(256) void gemm_bf16(const __bf16* __restrict__ A,
                                                 const __bf16* __restrict__ Bt,
                                                 const float* __restrict__ bias,
                                                 void* __restrict__ Cout,
                                                 int M, int N, int K,
                                                 long btHeadStride, long biasHeadStride) {
    __shared__ __attribute__((aligned(16))) __bf16 sA[2][128 * LDST];
    __shared__ __attribute__((aligned(16))) __bf16 sB[2][64 * LDST];

    const int h = blockIdx.z;
    const __bf16* Btz = Bt   + (size_t)h * btHeadStride;
    const float*  bz  = bias + (size_t)h * biasHeadStride;

    int wave    = threadIdx.x >> 5;
    int rowBase = blockIdx.x * 128 + wave * 16;
    int colBase = blockIdx.y * 64;
    const __bf16* Ablk = A   + (size_t)(blockIdx.x * 128) * K;
    const __bf16* Bblk = Btz + (size_t)colBase * K;

    // per-thread staging chunks: A = 512 x 16B, B = 256 x 16B
    int t   = threadIdx.x;
    int ca  = (t & 3) * 8;          // k-offset within 32-wide tile
    int ra0 = t >> 2;               // A rows 0..63
    int ra1 = ra0 + 64;             // A rows 64..127
    int rb  = t >> 2;               // B rows 0..63

    auto stage = [&](int k0, int bb) {
        __builtin_prefetch(Ablk + (size_t)ra0 * K + k0 + 64, 0, 0); // global_prefetch_b8
        async_copy16(Ablk + (size_t)ra0 * K + k0 + ca, &sA[bb][ra0 * LDST + ca]);
        async_copy16(Ablk + (size_t)ra1 * K + k0 + ca, &sA[bb][ra1 * LDST + ca]);
        async_copy16(Bblk + (size_t)rb  * K + k0 + ca, &sB[bb][rb  * LDST + ca]);
    };

    v8f acc[4] = {};
    const int nk = K / 32;
    stage(0, 0);
    for (int ks = 0; ks < nk; ++ks) {
        int cur = ks & 1;
        if (ks + 1 < nk) { stage((ks + 1) * 32, cur ^ 1); wait_async<3>(); }
        else             { wait_async<0>(); }
        __syncthreads();                         // all waves' tiles visible
        bf16x16 a = frag_lds(&sA[cur][wave * 16 * LDST]);
#pragma unroll
        for (int j = 0; j < 4; ++j) {
            bf16x16 b = frag_lds(&sB[cur][j * 16 * LDST]);
            acc[j] = wmma_bf16(a, b, acc[j]);
        }
        __syncthreads();                         // done reading before buffer reuse
    }

    // C/D layout: lane -> n = lane%16, VGPR v -> m = v + 8*(lane/16)
    int lane = threadIdx.x & 31;
    int n0   = lane & 15;
    int mofs = (lane >> 4) * 8;
#pragma unroll
    for (int j = 0; j < 4; ++j) {
#pragma unroll
        for (int v = 0; v < 8; ++v) {
            int row = rowBase + mofs + v;
            int col = colBase + j * 16 + n0;
            float val = acc[j][v] + bz[col];
            if constexpr (LAYOUT == 0) {
                ((__bf16*)Cout)[(size_t)row * N + col] = (__bf16)val;
            } else if constexpr (LAYOUT == 1) {
                int bb = row / Sq, s = row % Sq;
                int ns = col / DSd, d = col % DSd;
                size_t off = ((((size_t)h * Bn + bb) * NHs + ns) * Sq + s) * DSd + d;
                ((__bf16*)Cout)[off] = (__bf16)val;
            } else if constexpr (LAYOUT == 2) {
                int bb = row / Sq, s = row % Sq;
                int ns = col / DSd, d = col % DSd;
                size_t off = ((((size_t)h * Bn + bb) * NHs + ns) * DSd + d) * Sq + s;
                ((__bf16*)Cout)[off] = (__bf16)val;
            } else {
                ((float*)Cout)[(size_t)row * N + col] = val;
            }
        }
    }
}

// ---------------------------------------------------------------------------
// 4) scores[z] = SCALE * Qh[z] * Kh[z]^T (UNMASKED; mask applied in softmax)
//    z = (h*B+b)*NH+n ; grid = (S/128, S/64, 32)
// ---------------------------------------------------------------------------
__global__ __launch_bounds__(256) void attn_scores(const __bf16* __restrict__ Qh,
                                                   const __bf16* __restrict__ Kh,
                                                   float* __restrict__ attn) {
    int z = blockIdx.z;
    const __bf16* Qc = Qh + (size_t)z * Sq * DSd;
    const __bf16* Kc = Kh + (size_t)z * Sq * DSd;
    float*        Oc = attn + (size_t)z * Sq * Sq;

    int wave    = threadIdx.x >> 5;
    int rowBase = blockIdx.x * 128 + wave * 16;
    int colBase = blockIdx.y * 64;

    const __bf16* Arow = Qc + (size_t)rowBase * DSd;
    v8f acc[4] = {};
#pragma unroll
    for (int k0 = 0; k0 < DSd; k0 += 32) {
        bf16x16 a = load_frag_bf16(Arow, DSd, k0);
#pragma unroll
        for (int j = 0; j < 4; ++j) {
            bf16x16 bfr = load_frag_bf16(Kc + (size_t)(colBase + j * 16) * DSd, DSd, k0);
            acc[j] = wmma_bf16(a, bfr, acc[j]);
        }
    }

    int lane = threadIdx.x & 31;
    int n0   = lane & 15;
    int mofs = (lane >> 4) * 8;
#pragma unroll
    for (int j = 0; j < 4; ++j) {
#pragma unroll
        for (int v = 0; v < 8; ++v) {
            int row = rowBase + mofs + v;
            int col = colBase + j * 16 + n0;
            Oc[(size_t)row * Sq + col] = acc[j][v] * SCALE;
        }
    }
}

// ---------------------------------------------------------------------------
// 5) masked softmax, register-resident: one block per row (8 elems/thread);
//    attn touched exactly 1R + 1W. Mask from 1-bit bitmap.
// ---------------------------------------------------------------------------
__global__ __launch_bounds__(256) void softmax_rows(float* __restrict__ attn,
                                                    const unsigned* __restrict__ mbits) {
    __shared__ float sm[256];
    unsigned rowid = blockIdx.x;          // z*S + s
    int z = rowid >> 11;
    int s = rowid & (Sq - 1);
    int b = (z / NHs) % Bn;
    float* p = attn + (size_t)rowid * Sq;
    const unsigned* mrow = mbits + ((size_t)b * Sq + s) * (Sq / 32);
    int t = threadIdx.x;

    float v[8];
#pragma unroll
    for (int i = 0; i < 8; ++i) {
        int c = t + i * 256;
        float x = p[c];
        unsigned w = mrow[c >> 5];
        if (((w >> (c & 31)) & 1u) == 0u) x = -1e9f;
        v[i] = x;
    }
    float lmax = v[0];
#pragma unroll
    for (int i = 1; i < 8; ++i) lmax = fmaxf(lmax, v[i]);
    sm[t] = lmax; __syncthreads();
    for (int w2 = 128; w2 > 0; w2 >>= 1) {
        if (t < w2) sm[t] = fmaxf(sm[t], sm[t + w2]);
        __syncthreads();
    }
    float rmax = sm[0];
    __syncthreads();

    float lsum = 0.f;
#pragma unroll
    for (int i = 0; i < 8; ++i) { v[i] = __expf(v[i] - rmax); lsum += v[i]; }
    sm[t] = lsum; __syncthreads();
    for (int w2 = 128; w2 > 0; w2 >>= 1) {
        if (t < w2) sm[t] += sm[t + w2];
        __syncthreads();
    }
    float inv = 1.f / sm[0];
#pragma unroll
    for (int i = 0; i < 8; ++i) p[t + i * 256] = v[i] * inv;
}

// ---------------------------------------------------------------------------
// 6) ho[z] = attn[z] (fp32 -> bf16 in-reg) * Vh[z]; bf16 into merged mh[B,S,D]
// ---------------------------------------------------------------------------
__global__ __launch_bounds__(256) void attn_av(const float* __restrict__ attn,
                                               const __bf16* __restrict__ VhT,
                                               __bf16* __restrict__ mh) {
    int z  = blockIdx.z;
    int hN = z / (Bn * NHs);
    int b  = (z / NHs) % Bn;
    int n  = z % NHs;
    const float*  Ac = attn + (size_t)z * Sq * Sq;
    const __bf16* Vc = VhT  + (size_t)z * DSd * Sq;

    int wave    = threadIdx.x >> 5;
    int rowBase = blockIdx.x * 128 + wave * 16;

    const float* Arow = Ac + (size_t)rowBase * Sq;
    v8f acc[4] = {};
    for (int k0 = 0; k0 < Sq; k0 += 32) {
        __builtin_prefetch(Arow + k0 + 64, 0, 0);
        bf16x16 a = load_frag_f32_cvt(Arow, Sq, k0);
#pragma unroll
        for (int j = 0; j < 4; ++j) {
            bf16x16 bfr = load_frag_bf16(Vc + (size_t)(j * 16) * Sq, Sq, k0);
            acc[j] = wmma_bf16(a, bfr, acc[j]);
        }
    }

    int lane = threadIdx.x & 31;
    int n0   = lane & 15;
    int mofs = (lane >> 4) * 8;
#pragma unroll
    for (int j = 0; j < 4; ++j) {
#pragma unroll
        for (int v = 0; v < 8; ++v) {
            int row = rowBase + mofs + v;      // s
            int col = j * 16 + n0;             // d in [0,64)
            size_t off = ((size_t)b * Sq + row) * Dm + hN * DKd + n * DSd + col;
            mh[off] = (__bf16)acc[j][v];
        }
    }
}

// ---------------------------------------------------------------------------
// Host: stage the pipeline
// ---------------------------------------------------------------------------
extern "C" void kernel_launch(void* const* d_in, const int* in_sizes, int n_in,
                              void* d_out, int out_size, void* d_ws, size_t ws_size,
                              hipStream_t stream) {
    (void)in_sizes; (void)n_in; (void)out_size; (void)ws_size;

    const float* query  = (const float*)d_in[0];
    const float* key_in = (const float*)d_in[1];
    const float* value  = (const float*)d_in[2];
    const int*   mask   = (const int*)d_in[3];
    const float* Wq = (const float*)d_in[4];  const float* bq = (const float*)d_in[5];
    const float* Wk = (const float*)d_in[6];  const float* bk = (const float*)d_in[7];
    const float* Wv = (const float*)d_in[8];  const float* bv = (const float*)d_in[9];
    const float* Wh = (const float*)d_in[10]; const float* bh = (const float*)d_in[11];
    const float* Wo = (const float*)d_in[12]; const float* bo = (const float*)d_in[13];

    float* out  = (float*)d_out;                 // [B,S,D]
    float* attn = out + (size_t)BS * Dm;         // [4,B,NH,S,S] fp32

    // workspace carve-up (256B-aligned partitions)
    char* wsp = (char*)d_ws;
    auto takeB = [&](size_t bytes) -> char* {
        char* r = wsp;
        wsp += ((bytes + 255) & ~(size_t)255);
        return r;
    };
    auto take = [&](size_t elems) -> __bf16* { return (__bf16*)takeB(elems * sizeof(__bf16)); };

    __bf16* Xq  = take((size_t)BS * Dm);
    __bf16* Xk  = take((size_t)BS * Dm);
    __bf16* Xv  = take((size_t)BS * Dm);
    __bf16* WqT = take((size_t)Dm * Dm);
    __bf16* WkT = take((size_t)Dm * Dm);
    __bf16* WvT = take((size_t)Dm * Dm);
    __bf16* WoT = take((size_t)Dm * Dm);
    __bf16* WhT = take((size_t)Hn * DKd * Dm);
    __bf16* Qp  = take((size_t)BS * Dm);
    __bf16* Kp  = take((size_t)BS * Dm);
    __bf16* Vp  = take((size_t)BS * Dm);
    __bf16* Qh  = take((size_t)NCMB * Sq * DSd);
    __bf16* Kh  = take((size_t)NCMB * Sq * DSd);
    __bf16* VhT = take((size_t)NCMB * DSd * Sq);
    __bf16* mh  = take((size_t)BS * Dm);
    unsigned* mbits = (unsigned*)takeB((size_t)Bn * Sq * (Sq / 32) * sizeof(unsigned));

    // 0) activation converts + mask bit-pack
    {
        int n4 = (BS * Dm) / 4;
        dim3 g((n4 + 255) / 256), blk(256);
        cvt_f32_to_bf16<<<g, blk, 0, stream>>>(query,  Xq, n4);
        cvt_f32_to_bf16<<<g, blk, 0, stream>>>(key_in, Xk, n4);
        cvt_f32_to_bf16<<<g, blk, 0, stream>>>(value,  Xv, n4);
        pack_mask<<<dim3((Bn * Sq * Sq) / 256), blk, 0, stream>>>(mask, mbits);
    }
    // 1) weight transposes (-> Bt form)
    {
        dim3 blk(32, 8);
        dim3 gDD(Dm / 32, Dm / 32, 1);
        transpose_cvt<<<gDD, blk, 0, stream>>>(Wq, WqT, Dm, Dm, 0, 0);
        transpose_cvt<<<gDD, blk, 0, stream>>>(Wk, WkT, Dm, Dm, 0, 0);
        transpose_cvt<<<gDD, blk, 0, stream>>>(Wv, WvT, Dm, Dm, 0, 0);
        transpose_cvt<<<gDD, blk, 0, stream>>>(Wo, WoT, Dm, Dm, 0, 0);
        dim3 gH(DKd / 32, Dm / 32, Hn);
        transpose_cvt<<<gH, blk, 0, stream>>>(Wh, WhT, Dm, DKd,
                                              (long)Dm * DKd, (long)DKd * Dm);
    }
    // 2) shared projections: Qp/Kp/Vp = X @ W + b
    {
        dim3 blk(256), g(BS / 128, Dm / 64, 1);
        gemm_bf16<0><<<g, blk, 0, stream>>>(Xq, WqT, bq, Qp, BS, Dm, Dm, 0, 0);
        gemm_bf16<0><<<g, blk, 0, stream>>>(Xk, WkT, bk, Kp, BS, Dm, Dm, 0, 0);
        gemm_bf16<0><<<g, blk, 0, stream>>>(Xv, WvT, bv, Vp, BS, Dm, Dm, 0, 0);
    }
    // 3) per-named-head projections -> split sub-head layouts
    {
        dim3 blk(256), g(BS / 128, DKd / 64, Hn);
        gemm_bf16<1><<<g, blk, 0, stream>>>(Qp, WhT, bh, Qh, BS, DKd, Dm,
                                            (long)DKd * Dm, (long)DKd);
        gemm_bf16<1><<<g, blk, 0, stream>>>(Kp, WhT, bh, Kh, BS, DKd, Dm,
                                            (long)DKd * Dm, (long)DKd);
        gemm_bf16<2><<<g, blk, 0, stream>>>(Vp, WhT, bh, VhT, BS, DKd, Dm,
                                            (long)DKd * Dm, (long)DKd);
    }
    // 4) raw scaled scores into attn output region
    {
        dim3 blk(256), g(Sq / 128, Sq / 64, NCMB);
        attn_scores<<<g, blk, 0, stream>>>(Qh, Kh, attn);
    }
    // 5) masked softmax (register-resident, bitmap mask) -> final attn values
    {
        dim3 blk(256), g((unsigned)NCMB * Sq);
        softmax_rows<<<g, blk, 0, stream>>>(attn, mbits);
    }
    // 6) ho = attn @ Vh, merged directly into mh[B,S,D]
    {
        dim3 blk(256), g(Sq / 128, 1, NCMB);
        attn_av<<<g, blk, 0, stream>>>(attn, VhT, mh);
    }
    // 7) out = mh @ Wo + bo  (fp32 to d_out)
    {
        dim3 blk(256), g(BS / 128, Dm / 64, 1);
        gemm_bf16<3><<<g, blk, 0, stream>>>(mh, WoT, bo, out, BS, Dm, Dm, 0, 0);
    }
}
